// ViTLayer_37538014167630
// MI455X (gfx1250) — matrix operand order repaired and checked by reading, not compiled, not run on hardware
//
#include <hip/hip_runtime.h>
#include <stdint.h>

// Flip to 0 if the gfx1250 assembler rejects the async mnemonics.
#define USE_ASYNC_LDS 1

// ---------------------------------------------------------------------------
// Types for CDNA5 WMMA (wave32): bf16 inputs, f32 accumulate.
// ---------------------------------------------------------------------------
typedef __bf16 bf16_t;
typedef __attribute__((ext_vector_type(16))) __bf16 v16bf;
typedef __attribute__((ext_vector_type(8)))  float  v8f;

union V8F      { float f[8];  v8f  v; };
union BF16x16  { uint4 u[2];  v16bf v; };

__device__ __forceinline__ v8f v8f_zero() {
  V8F z;
#pragma unroll
  for (int i = 0; i < 8; ++i) z.f[i] = 0.f;
  return z.v;
}

// Async global->LDS copy (CDNA5 GLOBAL_LOAD_ASYNC_TO_LDS_B128, ASYNCcnt).
// ldsOff = wave-relative LDS byte address (low 32 bits of a generic pointer
// to __shared__ data on amdgcn); gptr = per-lane global source address.
__device__ __forceinline__ void async_copy_b128(uint32_t ldsOff, const void* gptr) {
#if USE_ASYNC_LDS
  asm volatile("global_load_async_to_lds_b128 %0, %1, off"
               :: "v"(ldsOff), "v"(gptr)
               : "memory");
#else
  uint4* d = (uint4*)(uintptr_t)0; (void)d; (void)ldsOff; (void)gptr;
#endif
}
__device__ __forceinline__ uint32_t lds_off(const void* p) {
  return (uint32_t)(uintptr_t)p;
}

// Problem dimensions
constexpr int Bdim  = 16;
constexpr int Sdim  = 1024;
constexpr int Ddim  = 768;
constexpr int Hdim  = 24;
constexpr int DHdim = 32;
constexpr int CHdim = 64;
constexpr int FFdim = 2304;
constexpr int Tdim  = Bdim * Sdim;   // 16384 tokens

// ---------------------------------------------------------------------------
// fp32 [K,N] -> bf16 [N,K] transpose+convert (weights).
// ---------------------------------------------------------------------------
__global__ void __launch_bounds__(256)
k_transpose_cvt(const float* __restrict__ W, bf16_t* __restrict__ Wt, int K, int N) {
  __shared__ float tile[32][33];
  const int k0 = blockIdx.x * 32;
  const int n0 = blockIdx.y * 32;
  for (int i = threadIdx.y; i < 32; i += 8)
    tile[i][threadIdx.x] = W[(size_t)(k0 + i) * N + (n0 + threadIdx.x)];
  __syncthreads();
  for (int i = threadIdx.y; i < 32; i += 8)
    Wt[(size_t)(n0 + i) * K + (k0 + threadIdx.x)] = (bf16_t)tile[threadIdx.x][i];
}

// ---------------------------------------------------------------------------
// LayerNorm (unbiased variance, ddof=1), one 256-thread block per token.
// ---------------------------------------------------------------------------
__global__ void __launch_bounds__(256)
k_ln(const float* __restrict__ x, const float* __restrict__ sc,
     const float* __restrict__ bi, bf16_t* __restrict__ out) {
  const int t   = blockIdx.x;
  const int tid = threadIdx.x;
  const float* row = x + (size_t)t * Ddim;
  const float v0 = row[tid], v1 = row[tid + 256], v2 = row[tid + 512];
  __shared__ float red[256];
  red[tid] = v0 + v1 + v2;
  __syncthreads();
  for (int o = 128; o > 0; o >>= 1) {
    if (tid < o) red[tid] += red[tid + o];
    __syncthreads();
  }
  const float mean = red[0] * (1.f / 768.f);
  __syncthreads();
  const float d0 = v0 - mean, d1 = v1 - mean, d2 = v2 - mean;
  red[tid] = d0 * d0 + d1 * d1 + d2 * d2;
  __syncthreads();
  for (int o = 128; o > 0; o >>= 1) {
    if (tid < o) red[tid] += red[tid + o];
    __syncthreads();
  }
  const float var  = red[0] * (1.f / 767.f);   // ddof=1
  const float rstd = rsqrtf(var + 1e-6f);
  bf16_t* orow = out + (size_t)t * Ddim;
  orow[tid]       = (bf16_t)(d0 * rstd * sc[tid]       + bi[tid]);
  orow[tid + 256] = (bf16_t)(d1 * rstd * sc[tid + 256] + bi[tid + 256]);
  orow[tid + 512] = (bf16_t)(d2 * rstd * sc[tid + 512] + bi[tid + 512]);
}

// ---------------------------------------------------------------------------
// WMMA GEMM core, double-buffered via async global->LDS copies.
// C[128x64] block tile, 8 waves x (16x64 strip, 4 accs), K stepped by 32.
// A bf16 row-major [M,K]; W bf16 transposed [N,K].
// Fragment addressing (contiguous k order, same in A and B):
//   A: lane = m + 16*h; K 16h..16h+15 contiguous
//   B: lanes 0-15: col n, K 0..15 ; lanes 16-31: col n, K 16..31
//   C: VGPR j -> row = 8*(lane>=16)+j, col = lane&15
// Pipeline per iteration: issue tile i+1 -> s_wait_asynccnt (tile i done) ->
// barrier -> WMMA on tile i -> barrier (buffer reuse safe).
// ---------------------------------------------------------------------------
template <bool DUAL>
__device__ __forceinline__ void gemm_core(
    const bf16_t* __restrict__ A, const bf16_t* __restrict__ W0,
    const bf16_t* __restrict__ W1, int K, int mBase, int nBase,
    bf16_t* As, bf16_t* Bs0, bf16_t* Bs1,   // As: 2*128*32, Bs*: 2*64*32
    v8f* acc0, v8f* acc1) {
  const int tid    = threadIdx.x;
  const int wave   = tid >> 5;
  const int lane   = tid & 31;
  const int laneLo = lane & 15;
  const int laneHi = lane >> 4;

  // per-thread staging coordinates
  const int am  = tid >> 1;          // A row (0..127), 16 elems each
  const int aks = (tid & 1) * 16;    // A k-segment
  const int bn  = tid >> 2;          // B row (0..63), 8 elems each
  const int bks = (tid & 3) * 8;     // B k-segment

#pragma unroll
  for (int t = 0; t < 4; ++t) {
    acc0[t] = v8f_zero();
    if (DUAL) acc1[t] = v8f_zero();
  }

  auto issue = [&](int kb, int p) {
    const bf16_t* asrc = A + (size_t)(mBase + am) * K + kb + aks;
    bf16_t*       adst = As + p * (128 * 32) + am * 32 + aks;
    async_copy_b128(lds_off(adst), asrc);
    async_copy_b128(lds_off(adst + 8), asrc + 8);
    async_copy_b128(lds_off(Bs0 + p * (64 * 32) + bn * 32 + bks),
                    W0 + (size_t)(nBase + bn) * K + kb + bks);
    if (DUAL)
      async_copy_b128(lds_off(Bs1 + p * (64 * 32) + bn * 32 + bks),
                      W1 + (size_t)(nBase + bn) * K + kb + bks);
  };

#if !USE_ASYNC_LDS
  auto stage_sync = [&](int kb, int p) {
    const uint4* src = reinterpret_cast<const uint4*>(A + (size_t)(mBase + am) * K + kb + aks);
    uint4* dst = reinterpret_cast<uint4*>(As + p * (128 * 32) + am * 32 + aks);
    dst[0] = src[0];
    dst[1] = src[1];
    reinterpret_cast<uint4*>(Bs0 + p * (64 * 32) + bn * 32 + bks)[0] =
        reinterpret_cast<const uint4*>(W0 + (size_t)(nBase + bn) * K + kb + bks)[0];
    if (DUAL)
      reinterpret_cast<uint4*>(Bs1 + p * (64 * 32) + bn * 32 + bks)[0] =
          reinterpret_cast<const uint4*>(W1 + (size_t)(nBase + bn) * K + kb + bks)[0];
  };
#endif

#if USE_ASYNC_LDS
  issue(0, 0);
#endif

  int p = 0;
  for (int kb = 0; kb < K; kb += 32, p ^= 1) {
    const bool hasNext = (kb + 32 < K);
#if USE_ASYNC_LDS
    if (hasNext) issue(kb + 32, p ^ 1);
    if (hasNext) {
      if (DUAL) asm volatile("s_wait_asynccnt 4" ::: "memory");
      else      asm volatile("s_wait_asynccnt 3" ::: "memory");
    } else {
      asm volatile("s_wait_asynccnt 0" ::: "memory");
    }
#else
    stage_sync(kb, p);
#endif
    __syncthreads();

    const bf16_t* Ab  = As  + p * (128 * 32);
    const bf16_t* B0b = Bs0 + p * (64 * 32);
    const bf16_t* B1b = Bs1 + p * (64 * 32);

    BF16x16 afrag;
    {
      const uint4* pa =
          reinterpret_cast<const uint4*>(Ab + (wave * 16 + laneLo) * 32 + laneHi * 16);
      afrag.u[0] = pa[0];   // K = 16h .. 16h+7
      afrag.u[1] = pa[1];   // K = 16h+8 .. 16h+15
    }
#pragma unroll
    for (int tn = 0; tn < 4; ++tn) {
      BF16x16 bfrag;
      const uint4* pb =
          reinterpret_cast<const uint4*>(B0b + (tn * 16 + laneLo) * 32 + laneHi * 16);
      bfrag.u[0] = pb[0];
      bfrag.u[1] = pb[1];
      acc0[tn] = __builtin_amdgcn_wmma_f32_16x16x32_bf16(
          false, afrag.v, false, bfrag.v, (short)0, acc0[tn], false, false);
      if (DUAL) {
        BF16x16 bfrag1;
        const uint4* pb1 =
            reinterpret_cast<const uint4*>(B1b + (tn * 16 + laneLo) * 32 + laneHi * 16);
        bfrag1.u[0] = pb1[0];
        bfrag1.u[1] = pb1[1];
        acc1[tn] = __builtin_amdgcn_wmma_f32_16x16x32_bf16(
            false, afrag.v, false, bfrag1.v, (short)0, acc1[tn], false, false);
      }
    }
    __syncthreads();   // buffer p may be overwritten next-next iteration
  }
}

// --- QKV projection: epilogue scatters to [B,H,S,DH] bf16 ------------------
__global__ void __launch_bounds__(256)
k_gemm_qkv(const bf16_t* __restrict__ A, const bf16_t* __restrict__ Wt,
           const float* __restrict__ bias, bf16_t* __restrict__ out) {
  __shared__ __align__(16) bf16_t As[2 * 128 * 32];
  __shared__ __align__(16) bf16_t Bs[2 * 64 * 32];
  v8f acc[4];
  const int mBase = blockIdx.x * 128;
  const int nBase = blockIdx.y * 64;
  gemm_core<false>(A, Wt, nullptr, Ddim, mBase, nBase, As, Bs, Bs, acc, nullptr);
  const int tid = threadIdx.x, wave = tid >> 5, lane = tid & 31;
  const int laneLo = lane & 15, laneHi = lane >> 4;
#pragma unroll
  for (int tn = 0; tn < 4; ++tn) {
    const int gn = nBase + tn * 16 + laneLo;
    const int h = gn >> 5, dh = gn & 31;
    const float bv = bias[gn];
#pragma unroll
    for (int j = 0; j < 8; ++j) {
      const int gm = mBase + wave * 16 + laneHi * 8 + j;
      const int b = gm >> 10, s = gm & 1023;
      out[(((size_t)b * Hdim + h) * Sdim + s) * DHdim + dh] = (bf16_t)(acc[tn][j] + bv);
    }
  }
}

// --- Wo projection: epilogue adds residual x -> x1 (f32) -------------------
__global__ void __launch_bounds__(256)
k_gemm_wo(const bf16_t* __restrict__ A, const bf16_t* __restrict__ Wt,
          const float* __restrict__ bias, const float* __restrict__ xin,
          float* __restrict__ x1) {
  __shared__ __align__(16) bf16_t As[2 * 128 * 32];
  __shared__ __align__(16) bf16_t Bs[2 * 64 * 32];
  v8f acc[4];
  const int mBase = blockIdx.x * 128;
  const int nBase = blockIdx.y * 64;
  gemm_core<false>(A, Wt, nullptr, Ddim, mBase, nBase, As, Bs, Bs, acc, nullptr);
  const int tid = threadIdx.x, wave = tid >> 5, lane = tid & 31;
  const int laneLo = lane & 15, laneHi = lane >> 4;
#pragma unroll
  for (int tn = 0; tn < 4; ++tn) {
    const int gn = nBase + tn * 16 + laneLo;
    const float bv = bias[gn];
#pragma unroll
    for (int j = 0; j < 8; ++j) {
      const int gm = mBase + wave * 16 + laneHi * 8 + j;
      x1[(size_t)gm * Ddim + gn] = acc[tn][j] + bv + xin[(size_t)gm * Ddim + gn];
    }
  }
}

// --- FFN1 fused: prod = (h@Wa+ba) * relu(h@Wb+bb) -> bf16 [T,FF] -----------
__global__ void __launch_bounds__(256)
k_gemm_ffn1(const bf16_t* __restrict__ A, const bf16_t* __restrict__ WtA,
            const bf16_t* __restrict__ WtB, const float* __restrict__ ba,
            const float* __restrict__ bb, bf16_t* __restrict__ prod) {
  __shared__ __align__(16) bf16_t As[2 * 128 * 32];
  __shared__ __align__(16) bf16_t Bs0[2 * 64 * 32];
  __shared__ __align__(16) bf16_t Bs1[2 * 64 * 32];
  v8f acc0[4], acc1[4];
  const int mBase = blockIdx.x * 128;
  const int nBase = blockIdx.y * 64;
  gemm_core<true>(A, WtA, WtB, Ddim, mBase, nBase, As, Bs0, Bs1, acc0, acc1);
  const int tid = threadIdx.x, wave = tid >> 5, lane = tid & 31;
  const int laneLo = lane & 15, laneHi = lane >> 4;
#pragma unroll
  for (int tn = 0; tn < 4; ++tn) {
    const int gn = nBase + tn * 16 + laneLo;
    const float bav = ba[gn], bbv = bb[gn];
#pragma unroll
    for (int j = 0; j < 8; ++j) {
      const int gm = mBase + wave * 16 + laneHi * 8 + j;
      const float a = acc0[tn][j] + bav;
      float g = acc1[tn][j] + bbv;
      g = g > 0.f ? g : 0.f;
      prod[(size_t)gm * FFdim + gn] = (bf16_t)(a * g);
    }
  }
}

// --- FFN2: out = x1 + prod@Wc + bc (f32) -----------------------------------
__global__ void __launch_bounds__(256)
k_gemm_ffn2(const bf16_t* __restrict__ A, const bf16_t* __restrict__ Wt,
            const float* __restrict__ bias, const float* __restrict__ x1,
            float* __restrict__ out) {
  __shared__ __align__(16) bf16_t As[2 * 128 * 32];
  __shared__ __align__(16) bf16_t Bs[2 * 64 * 32];
  v8f acc[4];
  const int mBase = blockIdx.x * 128;
  const int nBase = blockIdx.y * 64;
  gemm_core<false>(A, Wt, nullptr, FFdim, mBase, nBase, As, Bs, Bs, acc, nullptr);
  const int tid = threadIdx.x, wave = tid >> 5, lane = tid & 31;
  const int laneLo = lane & 15, laneHi = lane >> 4;
#pragma unroll
  for (int tn = 0; tn < 4; ++tn) {
    const int gn = nBase + tn * 16 + laneLo;
    const float bv = bias[gn];
#pragma unroll
    for (int j = 0; j < 8; ++j) {
      const int gm = mBase + wave * 16 + laneHi * 8 + j;
      out[(size_t)gm * Ddim + gn] = acc[tn][j] + bv + x1[(size_t)gm * Ddim + gn];
    }
  }
}

// ---------------------------------------------------------------------------
// LSH hash + stable counting sort. One block per (b,h).
// ---------------------------------------------------------------------------
__global__ void __launch_bounds__(256)
k_hash_sort(const bf16_t* __restrict__ q, const float* __restrict__ rot,
            int* __restrict__ idx) {
  const int h   = blockIdx.x;
  const int b   = blockIdx.y;
  const int tid = threadIdx.x;
  __shared__ float rotL[DHdim * 8];
  __shared__ int   histL[256 * 16];
  __shared__ int   totL[16];
  __shared__ int   baseL[16];

  if (tid < DHdim * 8) rotL[tid] = rot[(size_t)h * DHdim * 8 + tid];
#pragma unroll
  for (int c = 0; c < 16; ++c) histL[tid * 16 + c] = 0;
  __syncthreads();

  const bf16_t* qbh = q + ((size_t)(b * Hdim + h)) * Sdim * DHdim;
  int myCodes[4];
#pragma unroll
  for (int i = 0; i < 4; ++i) {
    const int s = tid * 4 + i;
    const bf16_t* qr = qbh + (size_t)s * DHdim;
    float p[8];
#pragma unroll
    for (int r = 0; r < 8; ++r) p[r] = 0.f;
    for (int d = 0; d < DHdim; ++d) {
      const float xv = (float)qr[d];
#pragma unroll
      for (int r = 0; r < 8; ++r) p[r] += xv * rotL[d * 8 + r];
    }
    int best = 0;
    float bv = p[0];
#pragma unroll
    for (int r = 1; r < 8; ++r)
      if (p[r] > bv) { bv = p[r]; best = r; }
#pragma unroll
    for (int r = 0; r < 8; ++r) {
      const float nv = -p[r];
      if (nv > bv) { bv = nv; best = 8 + r; }
    }
    myCodes[i] = best;
    histL[tid * 16 + best]++;
  }
  __syncthreads();

  if (tid < 16) {
    int tot = 0;
    for (int tt = 0; tt < 256; ++tt) tot += histL[tt * 16 + tid];
    totL[tid] = tot;
  }
  __syncthreads();
  if (tid == 0) {
    int run = 0;
    for (int c = 0; c < 16; ++c) { baseL[c] = run; run += totL[c]; }
  }
  __syncthreads();
  if (tid < 16) {
    int run = baseL[tid];
    for (int tt = 0; tt < 256; ++tt) {
      const int v2 = histL[tt * 16 + tid];
      histL[tt * 16 + tid] = run;
      run += v2;
    }
  }
  __syncthreads();

  int* idxbh = idx + ((size_t)(b * Hdim + h)) * Sdim;
#pragma unroll
  for (int i = 0; i < 4; ++i) {
    const int s   = tid * 4 + i;
    const int c   = myCodes[i];
    const int pos = histL[tid * 16 + c]++;
    idxbh[pos] = s;
  }
}

// ---------------------------------------------------------------------------
// Chunked attention: one 128-thread block (4 waves) per (chunk,h,b).
// q/k gathers use async global->LDS copies; v staged transposed.
// ---------------------------------------------------------------------------
__global__ void __launch_bounds__(128)
k_attn(const bf16_t* __restrict__ q, const bf16_t* __restrict__ kk,
       const bf16_t* __restrict__ vv, const int* __restrict__ idx,
       bf16_t* __restrict__ ctx) {
  const int c   = blockIdx.x;
  const int h   = blockIdx.y;
  const int b   = blockIdx.z;
  const int tid = threadIdx.x;
  const int wave = tid >> 5, lane = tid & 31;
  const int laneLo = lane & 15, laneHi = lane >> 4;

  __shared__ __align__(16) bf16_t qsL[64 * 32];
  __shared__ __align__(16) bf16_t ksL[64 * 32];
  __shared__ __align__(16) bf16_t vtL[32 * 64];   // transposed: [dh][row]
  __shared__ __align__(16) float  scL[64 * 64];
  __shared__ __align__(16) bf16_t atL[64 * 64];
  __shared__ int sArr[64];

  const size_t bh = (size_t)(b * Hdim + h);
  const int* idxc = idx + bh * Sdim + c * 64;
  const bf16_t* qbh = q  + bh * Sdim * DHdim;
  const bf16_t* kbh = kk + bh * Sdim * DHdim;
  const bf16_t* vbh = vv + bh * Sdim * DHdim;

  {  // gather: thread pair per row, 16 bf16 (32B) per thread
    const int r    = tid >> 1;
    const int half = tid & 1;
    const int sOrig = idxc[r];
    if (half == 0) sArr[r] = sOrig;
    const bf16_t* qp = qbh + (size_t)sOrig * DHdim + half * 16;
    const bf16_t* kp = kbh + (size_t)sOrig * DHdim + half * 16;
#if USE_ASYNC_LDS
    async_copy_b128(lds_off(qsL + r * 32 + half * 16), qp);
    async_copy_b128(lds_off(qsL + r * 32 + half * 16 + 8), qp + 8);
    async_copy_b128(lds_off(ksL + r * 32 + half * 16), kp);
    async_copy_b128(lds_off(ksL + r * 32 + half * 16 + 8), kp + 8);
#else
    reinterpret_cast<uint4*>(qsL + r * 32 + half * 16)[0] = reinterpret_cast<const uint4*>(qp)[0];
    reinterpret_cast<uint4*>(qsL + r * 32 + half * 16)[1] = reinterpret_cast<const uint4*>(qp)[1];
    reinterpret_cast<uint4*>(ksL + r * 32 + half * 16)[0] = reinterpret_cast<const uint4*>(kp)[0];
    reinterpret_cast<uint4*>(ksL + r * 32 + half * 16)[1] = reinterpret_cast<const uint4*>(kp)[1];
#endif
    const bf16_t* vp = vbh + (size_t)sOrig * DHdim + half * 16;
#pragma unroll
    for (int d = 0; d < 16; ++d) vtL[(half * 16 + d) * 64 + r] = vp[d];
  }
#if USE_ASYNC_LDS
  asm volatile("s_wait_asynccnt 0" ::: "memory");
#endif
  __syncthreads();

  const int mB = wave * 16;
  {  // scores = qs @ ks^T / sqrt(32):  M=64 N=64 K=32 (4 WMMAs/wave)
    BF16x16 aq;
    const uint4* pa = reinterpret_cast<const uint4*>(qsL + (mB + laneLo) * 32 + laneHi * 16);
    aq.u[0] = pa[0]; aq.u[1] = pa[1];
#pragma unroll
    for (int tn = 0; tn < 4; ++tn) {
      BF16x16 bk;
      const uint4* pb = reinterpret_cast<const uint4*>(ksL + (tn * 16 + laneLo) * 32 + laneHi * 16);
      bk.u[0] = pb[0]; bk.u[1] = pb[1];
      const v8f sc = __builtin_amdgcn_wmma_f32_16x16x32_bf16(
          false, aq.v, false, bk.v, (short)0, v8f_zero(), false, false);
#pragma unroll
      for (int j = 0; j < 8; ++j)
        scL[(mB + laneHi * 8 + j) * 64 + tn * 16 + laneLo] = sc[j] * 0.17677669529663687f;
    }
  }
  __syncthreads();

  if (tid < 64) {  // fp32 softmax over k, one row per thread
    float mx = -3.0e38f;
    for (int p2 = 0; p2 < 64; ++p2) mx = fmaxf(mx, scL[tid * 64 + p2]);
    float sum = 0.f;
    for (int p2 = 0; p2 < 64; ++p2) {
      const float e = __expf(scL[tid * 64 + p2] - mx);
      scL[tid * 64 + p2] = e;
      sum += e;
    }
    const float inv = 1.f / sum;
    for (int p2 = 0; p2 < 64; ++p2)
      atL[tid * 64 + p2] = (bf16_t)(scL[tid * 64 + p2] * inv);
  }
  __syncthreads();

  {  // o = att @ vs : M=64 N=32 K=64 (4 WMMAs/wave)
    v8f oacc[2];
    oacc[0] = v8f_zero();
    oacc[1] = v8f_zero();
#pragma unroll
    for (int kb = 0; kb < 64; kb += 32) {
      BF16x16 aa;
      const uint4* pa = reinterpret_cast<const uint4*>(atL + (mB + laneLo) * 64 + kb + laneHi * 16);
      aa.u[0] = pa[0]; aa.u[1] = pa[1];
#pragma unroll
      for (int tn = 0; tn < 2; ++tn) {
        BF16x16 bvv;
        const uint4* pb = reinterpret_cast<const uint4*>(vtL + (tn * 16 + laneLo) * 64 + kb + laneHi * 16);
        bvv.u[0] = pb[0]; bvv.u[1] = pb[1];
        oacc[tn] = __builtin_amdgcn_wmma_f32_16x16x32_bf16(
            false, aa.v, false, bvv.v, (short)0, oacc[tn], false, false);
      }
    }
#pragma unroll
    for (int tn = 0; tn < 2; ++tn) {
      const int dh = tn * 16 + laneLo;
#pragma unroll
      for (int j = 0; j < 8; ++j) {
        const int r = mB + laneHi * 8 + j;
        const int sOrig = sArr[r];
        ctx[((size_t)(b * Sdim + sOrig)) * Ddim + h * DHdim + dh] = (bf16_t)oacc[tn][j];
      }
    }
  }
}

// ---------------------------------------------------------------------------
// Launcher
// ---------------------------------------------------------------------------
extern "C" void kernel_launch(void* const* d_in, const int* in_sizes, int n_in,
                              void* d_out, int out_size, void* d_ws, size_t ws_size,
                              hipStream_t stream) {
  (void)in_sizes; (void)n_in; (void)out_size; (void)ws_size;

  const float* x    = (const float*)d_in[0];
  const float* ln1s = (const float*)d_in[1];
  const float* ln1b = (const float*)d_in[2];
  const float* Wq   = (const float*)d_in[3];
  const float* bq   = (const float*)d_in[4];
  const float* Wk   = (const float*)d_in[5];
  const float* bk   = (const float*)d_in[6];
  const float* Wv   = (const float*)d_in[7];
  const float* bvp  = (const float*)d_in[8];
  const float* Wo   = (const float*)d_in[9];
  const float* bo   = (const float*)d_in[10];
  const float* rot  = (const float*)d_in[11];
  const float* ln2s = (const float*)d_in[12];
  const float* ln2b = (const float*)d_in[13];
  const float* Wa   = (const float*)d_in[14];
  const float* ba   = (const float*)d_in[15];
  const float* Wb   = (const float*)d_in[16];
  const float* bb   = (const float*)d_in[17];
  const float* Wc   = (const float*)d_in[18];
  const float* bc   = (const float*)d_in[19];

  char* wsp = (char*)d_ws;
  size_t off = 0;
  auto take = [&](size_t bytes) -> void* {
    void* p = wsp + off;
    off += (bytes + 255) & ~(size_t)255;
    return p;
  };

  bf16_t* hbf = (bf16_t*)take((size_t)Tdim * Ddim * 2);
  bf16_t* WqT = (bf16_t*)take((size_t)Ddim * Ddim * 2);
  bf16_t* WkT = (bf16_t*)take((size_t)Ddim * Ddim * 2);
  bf16_t* WvT = (bf16_t*)take((size_t)Ddim * Ddim * 2);
  bf16_t* WoT = (bf16_t*)take((size_t)Ddim * Ddim * 2);
  bf16_t* WaT = (bf16_t*)take((size_t)FFdim * Ddim * 2);
  bf16_t* WbT = (bf16_t*)take((size_t)FFdim * Ddim * 2);
  bf16_t* WcT = (bf16_t*)take((size_t)Ddim * FFdim * 2);
  bf16_t* qb  = (bf16_t*)take((size_t)Bdim * Hdim * Sdim * DHdim * 2);
  bf16_t* kb2 = (bf16_t*)take((size_t)Bdim * Hdim * Sdim * DHdim * 2);
  bf16_t* vb2 = (bf16_t*)take((size_t)Bdim * Hdim * Sdim * DHdim * 2);
  int*    idx = (int*)take((size_t)Bdim * Hdim * Sdim * 4);
  bf16_t* ctx = (bf16_t*)take((size_t)Tdim * Ddim * 2);
  float*  x1  = (float*)take((size_t)Tdim * Ddim * 4);
  // FFN intermediate reuses the (dead-after-attention) q/k/v region:
  // 3 * 25,165,824 B == Tdim * FFdim * 2 B exactly, all 256-aligned.
  bf16_t* prod = qb;

  const dim3 tb(32, 8);
  k_transpose_cvt<<<dim3(Ddim / 32, Ddim / 32), tb, 0, stream>>>(Wq, WqT, Ddim, Ddim);
  k_transpose_cvt<<<dim3(Ddim / 32, Ddim / 32), tb, 0, stream>>>(Wk, WkT, Ddim, Ddim);
  k_transpose_cvt<<<dim3(Ddim / 32, Ddim / 32), tb, 0, stream>>>(Wv, WvT, Ddim, Ddim);
  k_transpose_cvt<<<dim3(Ddim / 32, Ddim / 32), tb, 0, stream>>>(Wo, WoT, Ddim, Ddim);
  k_transpose_cvt<<<dim3(Ddim / 32, FFdim / 32), tb, 0, stream>>>(Wa, WaT, Ddim, FFdim);
  k_transpose_cvt<<<dim3(Ddim / 32, FFdim / 32), tb, 0, stream>>>(Wb, WbT, Ddim, FFdim);
  k_transpose_cvt<<<dim3(FFdim / 32, Ddim / 32), tb, 0, stream>>>(Wc, WcT, FFdim, Ddim);

  k_ln<<<Tdim, 256, 0, stream>>>(x, ln1s, ln1b, hbf);

  const dim3 g768(Tdim / 128, Ddim / 64);
  k_gemm_qkv<<<g768, 256, 0, stream>>>(hbf, WqT, bq, qb);
  k_gemm_qkv<<<g768, 256, 0, stream>>>(hbf, WkT, bk, kb2);
  k_gemm_qkv<<<g768, 256, 0, stream>>>(hbf, WvT, bvp, vb2);

  k_hash_sort<<<dim3(Hdim, Bdim), 256, 0, stream>>>(qb, rot, idx);
  k_attn<<<dim3(Sdim / CHdim, Hdim, Bdim), 128, 0, stream>>>(qb, kb2, vb2, idx, ctx);

  k_gemm_wo<<<g768, 256, 0, stream>>>(ctx, WoT, bo, x, x1);

  k_ln<<<Tdim, 256, 0, stream>>>(x1, ln2s, ln2b, hbf);
  k_gemm_ffn1<<<dim3(Tdim / 128, FFdim / 64), 256, 0, stream>>>(hbf, WaT, WbT, ba, bb, prod);
  k_gemm_ffn2<<<g768, 256, 0, stream>>>(prod, WcT, bc, x1, (float*)d_out);
}
